// LinearAttention_expand_rel_SAM_75041668596127
// MI455X (gfx1250) — compile-verified, matrix operand order checked by reading, not run on hardware
//
#include <hip/hip_runtime.h>
#include <hip/hip_bf16.h>

// ---------------- types / WMMA helpers ----------------
typedef __attribute__((ext_vector_type(16))) _Float16 v16h;
typedef __attribute__((ext_vector_type(8)))  _Float16 v8h;
typedef __attribute__((ext_vector_type(4)))  _Float16 v4h;
typedef __attribute__((ext_vector_type(8)))  float    v8f;

__device__ inline v8f wmma_f16(v16h a, v16h b, v8f c) {
  // D = A(16x32 f16) * B(32x16 f16) + C(16x16 f32)
  return __builtin_amdgcn_wmma_f32_16x16x32_f16(
      /*neg_a=*/false, a, /*neg_b=*/false, b,
      /*c_mod=*/(short)0, c, /*reuse_a=*/false, /*reuse_b=*/false);
}

// A-fragment (16x32 f16) from a ROW-MAJOR [16][stride] buffer.
// ISA layout => lane(m,g) needs k in [g*8, g*8+8) and [16+g*8, 16+g*8+8):
// two contiguous 16B runs -> two b128 loads.
__device__ inline v16h load_a_frag_rm(const _Float16* __restrict__ src, int stride, int lane) {
  int m = lane & 15, g = lane >> 4;
  const _Float16* p = src + m * stride + g * 8;
  v8h lo = *(const v8h*)(p);
  v8h hi = *(const v8h*)(p + 16);
  return __builtin_shufflevector(lo, hi, 0, 1, 2, 3, 4, 5, 6, 7,
                                 8, 9, 10, 11, 12, 13, 14, 15);
}

// A-fragment from a ROW-MAJOR f32 row pointer (row already selected per lane).
__device__ inline v16h load_a_frag_f32(const float* __restrict__ row, int g) {
  const float4* p0 = (const float4*)(row + g * 8);
  const float4* p1 = (const float4*)(row + 16 + g * 8);
  float4 a = p0[0], b = p0[1], c = p1[0], d = p1[1];
  v16h r;
  r[0] = (_Float16)a.x;  r[1] = (_Float16)a.y;  r[2] = (_Float16)a.z;  r[3] = (_Float16)a.w;
  r[4] = (_Float16)b.x;  r[5] = (_Float16)b.y;  r[6] = (_Float16)b.z;  r[7] = (_Float16)b.w;
  r[8] = (_Float16)c.x;  r[9] = (_Float16)c.y;  r[10] = (_Float16)c.z; r[11] = (_Float16)c.w;
  r[12] = (_Float16)d.x; r[13] = (_Float16)d.y; r[14] = (_Float16)d.z; r[15] = (_Float16)d.w;
  return r;
}

// B-fragment from LDS staged in FRAGMENT ORDER (lane-major, 16 halves/lane):
// a single 32B read -> 2x ds_load_b128.
__device__ inline v16h load_b_frag_sw(const _Float16* __restrict__ fragbase, int lane) {
  return *(const v16h*)(fragbase + lane * 16);
}

// ---------------- problem constants ----------------
#define BATCH 32
#define DIM   256
#define HEADS 8
#define DH    64
#define INNER 512
#define HW    1024   // 32*32
#define NK    256    // 16*16
#define SCALE 0.125f

// ---------------- 1/5) depthwise 3x3 SAME, 8 outputs per thread ----------------
template <typename T>
__global__ __launch_bounds__(256) void dw3x3_kernel(const T* __restrict__ x,
                                                    const float* __restrict__ w,
                                                    _Float16* __restrict__ y,
                                                    int total8, int C) {
  int i = blockIdx.x * 256 + threadIdx.x;
  if (i >= total8) return;
  int w8 = i & 3;                 // 4 groups of 8 outputs per row
  int h = (i >> 2) & 31;
  long bc = (long)(i >> 7);
  int c = (int)(bc % C);
  const T* xb = x + bc * HW;
  const float* wt = w + (long)c * 9;
  int w0 = w8 * 8;
  float row[3][10];
#pragma unroll
  for (int dy = 0; dy < 3; ++dy) {
    int hh = h + dy - 1;
#pragma unroll
    for (int t = 0; t < 10; ++t) {
      int xx = w0 + t - 1;
      row[dy][t] = (hh >= 0 && hh < 32 && xx >= 0 && xx < 32)
                       ? (float)xb[hh * 32 + xx]
                       : 0.f;
    }
  }
  v8h out;
#pragma unroll
  for (int o = 0; o < 8; ++o) {
    float s = 0.f;
#pragma unroll
    for (int dy = 0; dy < 3; ++dy)
#pragma unroll
      for (int dx = 0; dx < 3; ++dx)
        s += row[dy][o + dx] * wt[dy * 3 + dx];
    out[o] = (_Float16)s;
  }
  *(v8h*)(y + bc * HW + h * 32 + w0) = out;
}

// ---------------- 2) pointwise qkv GEMM: (1536x256) @ (256 x B*1024) ----------------
// block: 256 thr = 8 waves; tile 128 oc x 64 p; K chunks of 32
// q/k/v destination is UNIFORM per block (128-oc tile never crosses a 512 boundary)
__global__ __launch_bounds__(256) void pw_qkv_gemm(const float* __restrict__ Wq,
                                                   const _Float16* __restrict__ X,
                                                   _Float16* __restrict__ qb,
                                                   _Float16* __restrict__ kf,
                                                   _Float16* __restrict__ vf) {
  __shared__ __align__(64) _Float16 Ws[128 * 32];  // row-major (A side)
  __shared__ __align__(64) _Float16 Xs[4 * 512];   // 4 B-frags in fragment order
  int tid = threadIdx.x, lane = tid & 31, wave = tid >> 5;
  int blk = blockIdx.x;
  int pt = blk & 15;
  int oct = (blk >> 4) % 12;
  int b = blk / (16 * 12);
  int ocbase = oct * 128, pbase = pt * 64;
  const _Float16* Xb = X + (long)b * DIM * HW;

  v8f zero = {};
  v8f acc[4];
#pragma unroll
  for (int nt = 0; nt < 4; ++nt) acc[nt] = zero;

  for (int kc = 0; kc < DIM; kc += 32) {
    __builtin_prefetch(&Wq[(long)ocbase * DIM + kc + 32], 0, 1);
    // Ws: float4 load -> v4h store (row-major)
    for (int gi = tid; gi < 128 * 8; gi += 256) {
      int rr = gi >> 3, c4 = gi & 7;
      float4 wv = *(const float4*)(Wq + (long)(ocbase + rr) * DIM + kc + c4 * 4);
      v4h hv;
      hv[0] = (_Float16)wv.x; hv[1] = (_Float16)wv.y;
      hv[2] = (_Float16)wv.z; hv[3] = (_Float16)wv.w;
      *(v4h*)(Ws + rr * 32 + c4 * 4) = hv;
    }
    // Xs: wide global load, swizzled b16 stores into fragment order
    {
      int k = tid >> 3, c8 = tid & 7;  // 32 x 8 groups == 256 threads
      v8h xv = *(const v8h*)(Xb + (long)(kc + k) * HW + pbase + c8 * 8);
      int g = (k >> 3) & 1;
      int e0 = (k & 7) + ((k & 16) ? 8 : 0);
#pragma unroll
      for (int t = 0; t < 8; ++t) {
        int col = c8 * 8 + t;
        int nt = col >> 4, n = col & 15;
        Xs[nt * 512 + (n + 16 * g) * 16 + e0] = xv[t];
      }
    }
    __syncthreads();
    v16h a = load_a_frag_rm(Ws + wave * 16 * 32, 32, lane);
#pragma unroll
    for (int nt = 0; nt < 4; ++nt) {
      v16h bf = load_b_frag_sw(Xs + nt * 512, lane);
      acc[nt] = wmma_f16(a, bf, acc[nt]);
    }
    __syncthreads();
  }

  // block-uniform q/k/v routing: grp = oct/4, sub-channel base = ocbase & 511
  _Float16* __restrict__ dst = (oct < 4) ? qb : ((oct < 8) ? kf : vf);
  int n = lane & 15, g = lane >> 4;
#pragma unroll
  for (int nt = 0; nt < 4; ++nt)
#pragma unroll
    for (int r = 0; r < 8; ++r) {
      int sub = (ocbase & 511) + wave * 16 + r + 8 * g;  // channel within q/k/v
      int p = pbase + nt * 16 + n;
      int d = sub >> 3, head = sub & 7;
      long idx = (((long)b * HEADS + head) * HW + p) * DH + d;
      dst[idx] = (_Float16)acc[nt][r];
    }
}

// ---------------- 3) 2x2 max pool, 8 channels per thread ----------------
__global__ __launch_bounds__(256) void pool2x2_kernel(const _Float16* __restrict__ kf,
                                                      const _Float16* __restrict__ vf,
                                                      _Float16* __restrict__ kp,
                                                      _Float16* __restrict__ vp) {
  const int half8 = BATCH * HEADS * NK * 8;  // d-groups of 8
  int i = blockIdx.x * 256 + threadIdx.x;
  if (i >= 2 * half8) return;
  int which = (i >= half8);
  int r = which ? (i - half8) : i;
  int d8 = r & 7;
  int t = r >> 3;
  int j = t & 255;
  long bh = (long)(t >> 8);
  int kh = j >> 4, kw = j & 15;
  const _Float16* src = (which ? vf : kf) + bh * HW * DH;
  int p00 = (2 * kh) * 32 + 2 * kw;
  v8h m0 = *(const v8h*)(src + (long)p00 * DH + d8 * 8);
  v8h m1 = *(const v8h*)(src + (long)(p00 + 1) * DH + d8 * 8);
  v8h m2 = *(const v8h*)(src + (long)(p00 + 32) * DH + d8 * 8);
  v8h m3 = *(const v8h*)(src + (long)(p00 + 33) * DH + d8 * 8);
  v8h out;
#pragma unroll
  for (int tt = 0; tt < 8; ++tt) {
    float m = fmaxf(fmaxf((float)m0[tt], (float)m1[tt]),
                    fmaxf((float)m2[tt], (float)m3[tt]));
    out[tt] = (_Float16)m;
  }
  *(v8h*)((which ? vp : kp) + (long)r * 8) = out;
}

// ---------------- 4) fused attention ----------------
// grid: B*HEADS*8; block: 256 thr = 8 waves; block tile: 128 q rows x 256 keys
__global__ __launch_bounds__(256) void attn_kernel(const _Float16* __restrict__ q,
                                                   const _Float16* __restrict__ kp,
                                                   const _Float16* __restrict__ vp,
                                                   const float* __restrict__ rel_pos_h,
                                                   const float* __restrict__ rel_pos_w,
                                                   float* __restrict__ attn_out,
                                                   _Float16* __restrict__ o) {
  // 32 KB: first K^T as 32 B-frags (2 c-chunks x 16 ntiles), then V as 32 B-frags
  __shared__ __align__(64) _Float16 KV[32 * 512];
  __shared__ __align__(64) float relh[128 * 16];  // 8 KB
  __shared__ __align__(64) float relw[128 * 16];  // 8 KB

  int tid = threadIdx.x, lane = tid & 31, wave = tid >> 5;
  int blk = blockIdx.x;
  int qblk = blk & 7;
  int bh = blk >> 3;
  int head = bh & 7, b = bh >> 3;
  int qbase = qblk * 128;

  const _Float16* qbh = q + (long)bh * HW * DH;
  const _Float16* kbh = kp + (long)bh * NK * DH;
  const _Float16* vbh = vp + (long)bh * NK * DH;
  float* abh = attn_out + (long)bh * HW * NK;

  // ---- stage K^T into fragment order: wide load + wide store ----
  for (int grp = tid; grp < 256 * 8; grp += 256) {   // (key j, c-group of 8)
    int j = grp >> 3, cg = grp & 7;
    v8h kv = *(const v8h*)(kbh + (long)j * DH + cg * 8);
    int c0 = cg * 8;
    int chunk = c0 >> 5;                // contraction chunk (c/32)
    int g = (c0 >> 3) & 1;
    int e0 = (c0 & 16) ? 8 : 0;
    int nt = j >> 4, n = j & 15;
    *(v8h*)(KV + (chunk * 16 + nt) * 512 + (n + 16 * g) * 16 + e0) = kv;
  }
  // ---- decomposed rel-pos biases (vectorized dot products) ----
  for (int i = tid; i < 128 * 16; i += 256) {
    int qr = i >> 4, kk = i & 15;
    int p = qbase + qr;
    int h = p >> 5, wq = p & 31;
    const _Float16* qrow = qbh + (long)p * DH;
    const float* rh = rel_pos_h + (long)(h - 2 * kk + 30) * DH;
    const float* rw = rel_pos_w + (long)(wq - 2 * kk + 30) * DH;
    float sh = 0.f, sw = 0.f;
#pragma unroll
    for (int c8 = 0; c8 < 8; ++c8) {
      v8h qv = *(const v8h*)(qrow + c8 * 8);
      const float4* h4 = (const float4*)(rh + c8 * 8);
      const float4* w4 = (const float4*)(rw + c8 * 8);
      float4 h0 = h4[0], h1 = h4[1], w0 = w4[0], w1 = w4[1];
      float q0 = (float)qv[0], q1 = (float)qv[1], q2 = (float)qv[2], q3 = (float)qv[3];
      float q4 = (float)qv[4], q5 = (float)qv[5], q6 = (float)qv[6], q7 = (float)qv[7];
      sh += q0 * h0.x + q1 * h0.y + q2 * h0.z + q3 * h0.w +
            q4 * h1.x + q5 * h1.y + q6 * h1.z + q7 * h1.w;
      sw += q0 * w0.x + q1 * w0.y + q2 * w0.z + q3 * w0.w +
            q4 * w1.x + q5 * w1.y + q6 * w1.z + q7 * w1.w;
    }
    relh[i] = sh;
    relw[i] = sw;
  }
  __syncthreads();

  int qrow0 = qbase + wave * 16;
  // Q A-frags straight from global (row-major, contiguous per lane)
  v16h aq0 = load_a_frag_rm(qbh + (long)qrow0 * DH, DH, lane);
  v16h aq1 = load_a_frag_rm(qbh + (long)qrow0 * DH + 32, DH, lane);

  v8f zero = {};
  v8f acc[16];
#pragma unroll
  for (int nt = 0; nt < 16; ++nt) acc[nt] = zero;

#pragma unroll
  for (int nt = 0; nt < 16; ++nt) {
    v16h b0 = load_b_frag_sw(KV + nt * 512, lane);
    v16h b1 = load_b_frag_sw(KV + (16 + nt) * 512, lane);
    acc[nt] = wmma_f16(aq0, b0, acc[nt]);
    acc[nt] = wmma_f16(aq1, b1, acc[nt]);
  }

  int n = lane & 15, g = lane >> 4;
  // bias (kh = nt, kw = n), scale AFTER bias as in reference; float4 LDS reads
#pragma unroll
  for (int r = 0; r < 8; ++r) {
    int qr = wave * 16 + r + 8 * g;
    const float4* rh4 = (const float4*)(relh + qr * 16);
    float rwv = relw[qr * 16 + n];
#pragma unroll
    for (int t = 0; t < 4; ++t) {
      float4 hb = rh4[t];
      acc[4 * t + 0][r] = (acc[4 * t + 0][r] + hb.x + rwv) * SCALE;
      acc[4 * t + 1][r] = (acc[4 * t + 1][r] + hb.y + rwv) * SCALE;
      acc[4 * t + 2][r] = (acc[4 * t + 2][r] + hb.z + rwv) * SCALE;
      acc[4 * t + 3][r] = (acc[4 * t + 3][r] + hb.w + rwv) * SCALE;
    }
  }

  // softmax over 256 keys
#pragma unroll
  for (int r = 0; r < 8; ++r) {
    float m = -3.4e38f;
#pragma unroll
    for (int nt = 0; nt < 16; ++nt) m = fmaxf(m, acc[nt][r]);
#pragma unroll
    for (int s = 1; s < 16; s <<= 1) m = fmaxf(m, __shfl_xor(m, s, 32));
    float sum = 0.f;
#pragma unroll
    for (int nt = 0; nt < 16; ++nt) {
      float e = __expf(acc[nt][r] - m);
      acc[nt][r] = e;
      sum += e;
    }
#pragma unroll
    for (int s = 1; s < 16; s <<= 1) sum += __shfl_xor(sum, s, 32);
    float inv = 1.f / sum;
#pragma unroll
    for (int nt = 0; nt < 16; ++nt) acc[nt][r] *= inv;
  }

  // write softmaxed attention (output #2)
#pragma unroll
  for (int nt = 0; nt < 16; ++nt)
#pragma unroll
    for (int r = 0; r < 8; ++r) {
      int row = qrow0 + r + 8 * g;
      abh[(long)row * NK + nt * 16 + n] = acc[nt][r];
    }
  __threadfence();
  __syncthreads();

  // ---- restage LDS with V in fragment order (8 key-chunks x 4 d-ntiles) ----
  for (int grp = tid; grp < 256 * 8; grp += 256) {   // (key j, c-group of 8)
    int j = grp >> 3, cg = grp & 7;
    v8h vv = *(const v8h*)(vbh + (long)j * DH + cg * 8);
    int chunk = j >> 5;
    int gg = (j >> 3) & 1;
    int e = (j & 7) + ((j & 16) ? 8 : 0);
#pragma unroll
    for (int t = 0; t < 8; ++t) {
      int c = cg * 8 + t;
      int nt = c >> 4, nn = c & 15;
      KV[(chunk * 4 + nt) * 512 + (nn + 16 * gg) * 16 + e] = vv[t];
    }
  }
  __syncthreads();

  // PV: out(16x64) = P(16x256) @ V(256x64); P re-read wide (f32 -> f16)
  v8f oacc[4];
#pragma unroll
  for (int nt = 0; nt < 4; ++nt) oacc[nt] = zero;
  const float* prow = abh + (long)(qrow0 + (lane & 15)) * NK;
#pragma unroll
  for (int kc = 0; kc < NK; kc += 32) {
    v16h ap = load_a_frag_f32(prow + kc, g);
#pragma unroll
    for (int nt = 0; nt < 4; ++nt) {
      v16h bv = load_b_frag_sw(KV + ((kc >> 5) * 4 + nt) * 512, lane);
      oacc[nt] = wmma_f16(ap, bv, oacc[nt]);
    }
  }

  // scatter to (B, INNER, 1024), channel = d*HEADS + head
#pragma unroll
  for (int nt = 0; nt < 4; ++nt)
#pragma unroll
    for (int r = 0; r < 8; ++r) {
      int d = nt * 16 + n;
      int row = qrow0 + r + 8 * g;
      int ch = d * HEADS + head;
      o[((long)b * INNER + ch) * HW + row] = (_Float16)oacc[nt][r];
    }
}

// ---------------- 6) pointwise out GEMM: (256x512) @ (512 x B*1024) ----------------
__global__ __launch_bounds__(256) void pw_out_gemm(const float* __restrict__ W2,
                                                   const _Float16* __restrict__ X,
                                                   float* __restrict__ Y) {
  __shared__ __align__(64) _Float16 Ws[128 * 32];
  __shared__ __align__(64) _Float16 Xs[4 * 512];
  int tid = threadIdx.x, lane = tid & 31, wave = tid >> 5;
  int blk = blockIdx.x;
  int pt = blk & 15;
  int oct = (blk >> 4) & 1;
  int b = blk >> 5;
  int ocbase = oct * 128, pbase = pt * 64;
  const _Float16* Xb = X + (long)b * INNER * HW;

  v8f zero = {};
  v8f acc[4];
#pragma unroll
  for (int nt = 0; nt < 4; ++nt) acc[nt] = zero;

  for (int kc = 0; kc < INNER; kc += 32) {
    __builtin_prefetch(&W2[(long)ocbase * INNER + kc + 32], 0, 1);
    for (int gi = tid; gi < 128 * 8; gi += 256) {
      int rr = gi >> 3, c4 = gi & 7;
      float4 wv = *(const float4*)(W2 + (long)(ocbase + rr) * INNER + kc + c4 * 4);
      v4h hv;
      hv[0] = (_Float16)wv.x; hv[1] = (_Float16)wv.y;
      hv[2] = (_Float16)wv.z; hv[3] = (_Float16)wv.w;
      *(v4h*)(Ws + rr * 32 + c4 * 4) = hv;
    }
    {
      int k = tid >> 3, c8 = tid & 7;
      v8h xv = *(const v8h*)(Xb + (long)(kc + k) * HW + pbase + c8 * 8);
      int g = (k >> 3) & 1;
      int e0 = (k & 7) + ((k & 16) ? 8 : 0);
#pragma unroll
      for (int t = 0; t < 8; ++t) {
        int col = c8 * 8 + t;
        int nt = col >> 4, n = col & 15;
        Xs[nt * 512 + (n + 16 * g) * 16 + e0] = xv[t];
      }
    }
    __syncthreads();
    v16h a = load_a_frag_rm(Ws + wave * 16 * 32, 32, lane);
#pragma unroll
    for (int nt = 0; nt < 4; ++nt) {
      v16h bf = load_b_frag_sw(Xs + nt * 512, lane);
      acc[nt] = wmma_f16(a, bf, acc[nt]);
    }
    __syncthreads();
  }

  int n = lane & 15, g = lane >> 4;
#pragma unroll
  for (int nt = 0; nt < 4; ++nt)
#pragma unroll
    for (int r = 0; r < 8; ++r) {
      int oc = ocbase + wave * 16 + r + 8 * g;
      int p = pbase + nt * 16 + n;
      Y[((long)b * DIM + oc) * HW + p] = acc[nt][r];
    }
}

// ---------------- launch ----------------
extern "C" void kernel_launch(void* const* d_in, const int* in_sizes, int n_in,
                              void* d_out, int out_size, void* d_ws, size_t ws_size,
                              hipStream_t stream) {
  const float* x      = (const float*)d_in[0];
  const float* dw_qkv = (const float*)d_in[1];
  const float* pw_qkv = (const float*)d_in[2];
  const float* dw_out = (const float*)d_in[3];
  const float* pw_out = (const float*)d_in[4];
  const float* rel_h  = (const float*)d_in[5];
  const float* rel_w  = (const float*)d_in[6];

  float* out  = (float*)d_out;                           // (B,256,32,32)
  float* attn = (float*)d_out + (long)BATCH * DIM * HW;  // (B,8,1024,256)

  char* ws = (char*)d_ws;
  _Float16* dwq = (_Float16*)(ws);
  _Float16* qb  = (_Float16*)(ws + (1L << 24));
  _Float16* kf  = (_Float16*)(ws + (1L << 24) + (1L << 25));
  _Float16* vf  = (_Float16*)(ws + (1L << 24) + 2 * (1L << 25));
  _Float16* kpb = (_Float16*)(ws + (1L << 24) + 3 * (1L << 25));
  _Float16* vpb = (_Float16*)(ws + (1L << 24) + 3 * (1L << 25) + (1L << 23));
  _Float16* o   = (_Float16*)(ws + (1L << 24) + 3 * (1L << 25) + 2 * (1L << 23));
  _Float16* od  = (_Float16*)(ws + (1L << 24) + 4 * (1L << 25) + 2 * (1L << 23));

  // 1) depthwise 3x3 on x (256 ch), f32 -> f16
  {
    int total8 = BATCH * DIM * 32 * 4;
    dw3x3_kernel<float><<<(total8 + 255) / 256, 256, 0, stream>>>(x, dw_qkv, dwq, total8, DIM);
  }
  // 2) pointwise qkv GEMM with head-major scatter
  pw_qkv_gemm<<<BATCH * 12 * 16, 256, 0, stream>>>(pw_qkv, dwq, qb, kf, vf);
  // 3) pool k, v
  {
    int total = 2 * BATCH * HEADS * NK * 8;
    pool2x2_kernel<<<(total + 255) / 256, 256, 0, stream>>>(kf, vf, kpb, vpb);
  }
  // 4) fused attention (writes attn output + o)
  attn_kernel<<<BATCH * HEADS * 8, 256, 0, stream>>>(qb, kpb, vpb, rel_h, rel_w, attn, o);
  // 5) depthwise 3x3 on o (512 ch), f16 -> f16
  {
    int total8 = BATCH * INNER * 32 * 4;
    dw3x3_kernel<_Float16><<<(total8 + 255) / 256, 256, 0, stream>>>(o, dw_out, od, total8, INNER);
  }
  // 6) pointwise out GEMM -> f32 output
  pw_out_gemm<<<BATCH * 2 * 16, 256, 0, stream>>>(pw_out, od, out);
}
